// GATConv_10977936409071
// MI455X (gfx1250) — matrix operand order repaired and checked by reading, not compiled, not run on hardware
//
#include <hip/hip_runtime.h>

#define BB 8
#define NN 4096
#define DD 64
#define SLOPE 0.01f
#define MASKV (-3.0e38f)

typedef __bf16 bf16_t;
typedef __attribute__((ext_vector_type(16))) __bf16 v16bf;
typedef __attribute__((ext_vector_type(8)))  __bf16 v8bf;
typedef __attribute__((ext_vector_type(8)))  float  v8f;

union AF { v16bf v; v8bf h[2]; bf16_t e[16]; };
union F8 { v8f v; float e[8]; };

static __device__ __forceinline__ v8f wmma_bf16(v16bf a, v16bf b, v8f c) {
  // D = A(16x32 bf16) * B(32x16 bf16) + C(16x16 f32)
  return __builtin_amdgcn_wmma_f32_16x16x32_bf16(false, a, false, b, (short)0, c,
                                                 false, false);
}

static __device__ __forceinline__ AF ldfrag(const bf16_t* p) {
  AF f;
  f.h[0] = *(const v8bf*)(p);
  f.h[1] = *(const v8bf*)(p + 16);
  return f;
}

// ---------------------------------------------------------------------------
// Kernel 0: pack adjacency (int 0/1) into a bitmask, self-loops baked in.
// 64 MB -> 2 MB; read once from HBM, then L2-resident for the attention pass.
// ---------------------------------------------------------------------------
__global__ __launch_bounds__(256) void gat_pack_adj(const int* __restrict__ Adj,
                                                    unsigned* __restrict__ Bits)
{
  const int w   = blockIdx.x * 256 + threadIdx.x;  // 0 .. N*N/32-1
  const int row = w >> 7;                          // N/32 = 128 words per row
  const int wc  = w & 127;
  const int cb  = wc * 32;
  const int* src = Adj + (size_t)row * NN + cb;
  unsigned bits = 0u;
#pragma unroll
  for (int j = 0; j < 32; ++j) bits |= (src[j] > 0 ? 1u : 0u) << j;
  if (row >= cb && row < cb + 32) bits |= 1u << (row - cb);  // self-loop
  Bits[w] = bits;
}

// ---------------------------------------------------------------------------
// Kernel 1: projections  q = X@Wq, k = X@Wk, v = X@Wv  (v stored transposed)
// One wave handles one 16-row tile of one matrix. bf16 outputs.
// ---------------------------------------------------------------------------
__global__ __launch_bounds__(128) void gat_proj_kernel(
    const float* __restrict__ X,  const float* __restrict__ W0,
    const float* __restrict__ W1, const float* __restrict__ W2,
    bf16_t* __restrict__ qb, bf16_t* __restrict__ kb, bf16_t* __restrict__ vT)
{
  const int lane = threadIdx.x & 31;
  const int wv   = threadIdx.x >> 5;
  const int gt   = blockIdx.x * 4 + wv;      // 0 .. B*N/16-1
  const int b    = gt >> 8;                  // N/16 == 256 tiles per batch
  const int rt   = gt & 255;
  const int mat  = blockIdx.y;               // 0: W->vT, 1: Q->q, 2: K->k
  const int hi   = lane >> 4;
  const int l16  = lane & 15;

  const float* Wm   = (mat == 0) ? W0 : (mat == 1 ? W1 : W2);
  bf16_t*      dst  = (mat == 0) ? vT : (mat == 1 ? qb : kb);
  const bool   tr   = (mat == 0);

  // A-fragments: rows rt*16..+15 of X, K=0..63 in two 16x32 frags.
  const float* xrow = X + ((size_t)b * NN + rt * 16 + l16) * DD;
  const int c0 = 8 * hi;
  AF a0, a1;
#pragma unroll
  for (int e = 0; e < 8; ++e) {
    a0.e[e]     = (bf16_t)xrow[c0 + e];
    a0.e[e + 8] = (bf16_t)xrow[c0 + 16 + e];
    a1.e[e]     = (bf16_t)xrow[32 + c0 + e];
    a1.e[e + 8] = (bf16_t)xrow[48 + c0 + e];
  }

#pragma unroll
  for (int t = 0; t < 4; ++t) {
    AF bw0, bw1;
#pragma unroll
    for (int e = 0; e < 16; ++e) {
      bw0.e[e] = (bf16_t)Wm[(8 * hi + e + (e & 8)) * DD + t * 16 + l16];
      bw1.e[e] = (bf16_t)Wm[(32 + 8 * hi + e + (e & 8)) * DD + t * 16 + l16];
    }
    F8 acc; acc.v = (v8f){0.f, 0.f, 0.f, 0.f, 0.f, 0.f, 0.f, 0.f};
    acc.v = wmma_bf16(a0.v, bw0.v, acc.v);
    acc.v = wmma_bf16(a1.v, bw1.v, acc.v);
#pragma unroll
    for (int r = 0; r < 8; ++r) {
      const int row = rt * 16 + r + 8 * hi;   // C/D layout: M = r + 8*hi
      const int col = t * 16 + l16;           //             N = l16
      const size_t idxT = ((size_t)b * DD + col) * NN + row;   // transposed (vT)
      const size_t idxR = ((size_t)b * NN + row) * DD + col;   // row-major (q,k)
      dst[tr ? idxT : idxR] = (bf16_t)acc.e[r];                // uniform select
    }
  }
}

// ---------------------------------------------------------------------------
// Kernel 2: fused flash-attention with leaky_relu + bitmask adjacency.
// One wave per 16-query-row tile; streams columns in chunks of 64.
// ---------------------------------------------------------------------------
__global__ __launch_bounds__(128) void gat_attn_kernel(
    const bf16_t* __restrict__ qb, const bf16_t* __restrict__ kb,
    const bf16_t* __restrict__ vT, const unsigned* __restrict__ Bits,
    float* __restrict__ out)
{
  __shared__ __align__(16) bf16_t lds_p[4][16 * 64];

  const int lane = threadIdx.x & 31;
  const int wv   = threadIdx.x >> 5;
  const int gt   = blockIdx.x * 4 + wv;
  const int b    = gt >> 8;
  const int rt   = gt & 255;
  const int hi   = lane >> 4;
  const int l16  = lane & 15;
  const int m0   = rt * 16;
  const int c0   = 8 * hi;

  // Persistent q A-fragments (K = 0..31 and 32..63)
  const bf16_t* qrow = qb + ((size_t)b * NN + m0 + l16) * DD;
  AF aq0, aq1;
  aq0.h[0] = *(const v8bf*)(qrow + c0);
  aq0.h[1] = *(const v8bf*)(qrow + c0 + 16);
  aq1.h[0] = *(const v8bf*)(qrow + 32 + c0);
  aq1.h[1] = *(const v8bf*)(qrow + 48 + c0);

  float mrow[8], lrow[8];
  F8 acco[4];
  const unsigned* brow[8];
#pragma unroll
  for (int r = 0; r < 8; ++r) {
    mrow[r] = MASKV;
    lrow[r] = 0.f;
    brow[r] = Bits + (size_t)(m0 + r + 8 * hi) * (NN / 32);
  }
#pragma unroll
  for (int t = 0; t < 4; ++t)
    acco[t].v = (v8f){0.f, 0.f, 0.f, 0.f, 0.f, 0.f, 0.f, 0.f};

  bf16_t* lds = &lds_p[wv][0];
  const bf16_t* kbase = kb + (size_t)b * NN * DD;
  const bf16_t* vbase = vT + (size_t)b * DD * NN;

  for (int cb = 0; cb < NN; cb += 64) {
    if (cb + 64 < NN)  // warm caches for the next step
      __builtin_prefetch(kbase + (size_t)(cb + 64 + l16) * DD, 0, 1);

    // ---- S = Q K^T for a 16x64 block (K-dim = 64) ----
    F8 s[4];
#pragma unroll
    for (int nt = 0; nt < 4; ++nt) {
      const bf16_t* kp = kbase + (size_t)(cb + nt * 16 + l16) * DD + 8 * hi;
      AF bka = ldfrag(kp);        // K = 0..31
      AF bkb = ldfrag(kp + 32);   // K = 32..63
      s[nt].v = (v8f){0.f, 0.f, 0.f, 0.f, 0.f, 0.f, 0.f, 0.f};
      s[nt].v = wmma_bf16(aq0.v, bka.v, s[nt].v);
      s[nt].v = wmma_bf16(aq1.v, bkb.v, s[nt].v);
    }

    // ---- leaky_relu + bitmask + online softmax ----
#pragma unroll
    for (int r = 0; r < 8; ++r) {
      const uint2 wb = *(const uint2*)(brow[r] + (cb >> 5));  // 64 column bits
      const bool k0 = (wb.x >> l16) & 1u;
      const bool k1 = (wb.x >> (16 + l16)) & 1u;
      const bool k2 = (wb.y >> l16) & 1u;
      const bool k3 = (wb.y >> (16 + l16)) & 1u;

      float v0 = s[0].e[r]; v0 = (v0 > 0.f) ? v0 : SLOPE * v0; v0 = k0 ? v0 : MASKV;
      float v1 = s[1].e[r]; v1 = (v1 > 0.f) ? v1 : SLOPE * v1; v1 = k1 ? v1 : MASKV;
      float v2 = s[2].e[r]; v2 = (v2 > 0.f) ? v2 : SLOPE * v2; v2 = k2 ? v2 : MASKV;
      float v3 = s[3].e[r]; v3 = (v3 > 0.f) ? v3 : SLOPE * v3; v3 = k3 ? v3 : MASKV;

      float rmax = fmaxf(fmaxf(v0, v1), fmaxf(v2, v3));
#pragma unroll
      for (int msk = 1; msk <= 8; msk <<= 1)
        rmax = fmaxf(rmax, __shfl_xor(rmax, msk, 32));  // stays within 16-lane half
      const float mnew = fmaxf(mrow[r], rmax);

      float p0 = __expf(v0 - mnew); p0 = k0 ? p0 : 0.f;
      float p1 = __expf(v1 - mnew); p1 = k1 ? p1 : 0.f;
      float p2 = __expf(v2 - mnew); p2 = k2 ? p2 : 0.f;
      float p3 = __expf(v3 - mnew); p3 = k3 ? p3 : 0.f;

      float rs = (p0 + p1) + (p2 + p3);
#pragma unroll
      for (int msk = 1; msk <= 8; msk <<= 1)
        rs += __shfl_xor(rs, msk, 32);

      const float corr = __expf(mrow[r] - mnew);
      lrow[r] = lrow[r] * corr + rs;
      mrow[r] = mnew;
#pragma unroll
      for (int t = 0; t < 4; ++t) acco[t].e[r] *= corr;

      bf16_t* lrow_p = lds + (r + 8 * hi) * 64 + l16;   // C/D -> LDS (row-major 16x64)
      lrow_p[0]  = (bf16_t)p0;
      lrow_p[16] = (bf16_t)p1;
      lrow_p[32] = (bf16_t)p2;
      lrow_p[48] = (bf16_t)p3;
    }

    // P back as two 16x32 A-fragments (cols cb..cb+31 and cb+32..cb+63)
    AF ap0, ap1;
    ap0.h[0] = *(const v8bf*)(lds + l16 * 64 + c0);
    ap0.h[1] = *(const v8bf*)(lds + l16 * 64 + 16 + c0);
    ap1.h[0] = *(const v8bf*)(lds + l16 * 64 + 32 + c0);
    ap1.h[1] = *(const v8bf*)(lds + l16 * 64 + 48 + c0);

    // ---- O += P * V  (V pre-transposed: contiguous K per lane) ----
#pragma unroll
    for (int t = 0; t < 4; ++t) {
      const bf16_t* vp = vbase + (size_t)(t * 16 + l16) * NN + cb + 8 * hi;
      AF bv0 = ldfrag(vp);        // K rows cb    .. cb+31
      AF bv1 = ldfrag(vp + 32);   // K rows cb+32 .. cb+63
      acco[t].v = wmma_bf16(ap0.v, bv0.v, acco[t].v);
      acco[t].v = wmma_bf16(ap1.v, bv1.v, acco[t].v);
    }
  }

  // ---- epilogue: normalize and store fp32 ----
  float inv[8];
#pragma unroll
  for (int r = 0; r < 8; ++r) inv[r] = 1.0f / lrow[r];
#pragma unroll
  for (int t = 0; t < 4; ++t) {
#pragma unroll
    for (int r = 0; r < 8; ++r) {
      const int grow = m0 + r + 8 * hi;
      out[((size_t)b * NN + grow) * DD + t * 16 + l16] = acco[t].e[r] * inv[r];
    }
  }
}

// ---------------------------------------------------------------------------
extern "C" void kernel_launch(void* const* d_in, const int* in_sizes, int n_in,
                              void* d_out, int out_size, void* d_ws, size_t ws_size,
                              hipStream_t stream) {
  (void)in_sizes; (void)n_in; (void)out_size; (void)ws_size;
  const float* X      = (const float*)d_in[0];
  const int*   Adj    = (const int*)d_in[1];
  const float* weight = (const float*)d_in[2];
  const float* query  = (const float*)d_in[3];
  const float* key_p  = (const float*)d_in[4];
  float* out = (float*)d_out;

  const size_t elems = (size_t)BB * NN * DD;
  bf16_t*   qb   = (bf16_t*)d_ws;
  bf16_t*   kb   = qb + elems;
  bf16_t*   vT   = kb + elems;
  unsigned* Bits = (unsigned*)(vT + elems);   // N*N/32 words = 2 MB

  gat_pack_adj<<<dim3(NN * (NN / 32) / 256), 256, 0, stream>>>(Adj, Bits);

  dim3 g1(BB * NN / 64, 3);   // 4 waves per block, one 16-row tile per wave
  gat_proj_kernel<<<g1, 128, 0, stream>>>(X, weight, query, key_p, qb, kb, vT);

  dim3 g2(BB * NN / 64);
  gat_attn_kernel<<<g2, 128, 0, stream>>>(qb, kb, vT, Bits, out);
}
